// NeuralSDF_32736240730809
// MI455X (gfx1250) — matrix operand order repaired and checked by reading, not compiled, not run on hardware
//
#include <hip/hip_runtime.h>

// Fused NeuralSDF forward + input-gradient kernel for gfx1250 (MI455X).
// 128 threads (4 wave32) per workgroup, 16 samples per wave (64 per WG).
// MLP GEMMs run on v_wmma_f32_16x16x32_f16 (f32 accumulate).
// LDS uses padded row strides (136 halfs = 68 dwords = 4 mod 64 banks) so all
// fragment loads are conflict-free ds_load_b128 with immediate offsets.

typedef __attribute__((ext_vector_type(16))) _Float16 v16h;
typedef __attribute__((ext_vector_type(8)))  float    v8f;

union FragU { v16h v; float4 q[2]; _Float16 h[16]; };

#define SWW 136   // row stride (halfs) for 128-wide LDS buffers
#define SWF 40    // row stride (halfs) for 32-wide LDS buffers

__device__ __forceinline__ int offW(int row, int k){ return row*SWW + k; }
__device__ __forceinline__ int offF(int row, int k){ return row*SWF + k; }

__device__ __forceinline__ v8f wmma_f16(const FragU& a, const FragU& b, v8f c){
  return __builtin_amdgcn_wmma_f32_16x16x32_f16(false, a.v, false, b.v,
                                                (short)0, c, false, false);
}

#define WAVES_PER_BLOCK 4

__global__ __launch_bounds__(32 * WAVES_PER_BLOCK) void neural_sdf_fused(
    const float* __restrict__ positions,
    const float* __restrict__ encGrid,        // 256^3 x 8
    const unsigned char* __restrict__ mask,   // 64^3 bool
    const float* __restrict__ emptyGrid,      // 4^3 x 8
    const float* __restrict__ W1g, const float* __restrict__ b1g,  // 11x128, 128
    const float* __restrict__ W2g, const float* __restrict__ b2g,  // 128x128, 128
    const float* __restrict__ W3g, const float* __restrict__ b3g,  // 128x1, 1
    float* __restrict__ outSdf, float* __restrict__ outGrad, int N)
{
  // ---- LDS: ~128.8 KB total (2 WGs fit in a 320 KB WGP) ----
  __shared__ __align__(16) _Float16 sW1T[128*SWF];              // sW1T[n][k] = W1[k][n]  (fwd L1 B)
  __shared__ __align__(16) _Float16 sW1 [16*SWW];               // sW1[k][i]  = W1[k][i]  (bwd L1 B), rows 11..15 = 0
  __shared__ __align__(16) _Float16 sW2T[128*SWW];              // sW2T[j][i] = W2[i][j]  (fwd L2 B)
  __shared__ __align__(16) _Float16 sW2 [128*SWW];              // sW2[i][j]  = W2[i][j]  (bwd L2 B)
  __shared__ float sW3[128];
  __shared__ float sB1[128];
  __shared__ float sB2[128];
  __shared__ __align__(16) _Float16 sFeat[WAVES_PER_BLOCK][16*SWF]; // later reused as f32 red[16][16]
  __shared__ __align__(16) _Float16 sZ1 [WAVES_PER_BLOCK][16*SWW]; // z1, overwritten in-place by gz1
  __shared__ __align__(16) _Float16 sGz2[WAVES_PER_BLOCK][16*SWW];
  __shared__ float sDemb[WAVES_PER_BLOCK][16*24];               // [sample][dim(3)][ch(8)]

  const int tid  = threadIdx.x;
  const int wave = tid >> 5;
  const int lane = tid & 31;
  const int waveSample0 = blockIdx.x * (16 * WAVES_PER_BLOCK) + wave * 16;

  _Float16* featB = &sFeat[wave][0];
  _Float16* z1B   = &sZ1 [wave][0];
  _Float16* gz2B  = &sGz2[wave][0];
  float*    dembB = &sDemb[wave][0];
  float*    redB  = reinterpret_cast<float*>(featB);  // reused after L1 fwd

  // ---- phase 1: trilerp gather (2 lanes per sample, 4 channels each) ----
  // Issued first so the random 16B gathers overlap the weight staging below.
  {
    const int s = lane >> 1, h = lane & 1;
    const int gs = waveSample0 + s;
    const float px = positions[gs*3+0];
    const float py = positions[gs*3+1];
    const float pz = positions[gs*3+2];

    int bx = min(max((int)floorf(px*64.f), 0), 63);
    int by = min(max((int)floorf(py*64.f), 0), 63);
    int bz = min(max((int)floorf(pz*64.f), 0), 63);
    const bool valid = mask[(bx*64 + by)*64 + bz] != 0;

    const float* G; float bmin, scale; int R;
    if (valid){ G = encGrid;   bmin = 0.0f;  scale = 255.0f;     R = 256; }
    else      { G = emptyGrid; bmin = -0.1f; scale = 3.0f/1.2f;  R = 4;   }

    const float ux = (px - bmin)*scale, uy = (py - bmin)*scale, uz = (pz - bmin)*scale;
    const int ix = min(max((int)floorf(ux), 0), R-2);
    const int iy = min(max((int)floorf(uy), 0), R-2);
    const int iz = min(max((int)floorf(uz), 0), R-2);
    const float fx = ux - (float)ix, fy = uy - (float)iy, fz = uz - (float)iz;

    float e[4]  = {0,0,0,0};
    float dX[4] = {0,0,0,0}, dY[4] = {0,0,0,0}, dZ[4] = {0,0,0,0};
    #pragma unroll
    for (int dx = 0; dx < 2; ++dx){
      const float wx = dx ? fx : 1.f - fx;  const float gx = dx ? 1.f : -1.f;
      #pragma unroll
      for (int dy = 0; dy < 2; ++dy){
        const float wy = dy ? fy : 1.f - fy;  const float gy = dy ? 1.f : -1.f;
        #pragma unroll
        for (int dz = 0; dz < 2; ++dz){
          const float wz = dz ? fz : 1.f - fz;  const float gz = dz ? 1.f : -1.f;
          const float* cp = G + ((((ix+dx)*R + (iy+dy))*R + (iz+dz)) << 3) + (h << 2);
          const float4 c4 = *(const float4*)cp;
          const float cc[4] = {c4.x, c4.y, c4.z, c4.w};
          const float w  = wx*wy*wz;
          const float ax = gx*wy*wz, ay = wx*gy*wz, az = wx*wy*gz;
          #pragma unroll
          for (int q = 0; q < 4; ++q){
            e[q]  += w  * cc[q];
            dX[q] += ax * cc[q];
            dY[q] += ay * cc[q];
            dZ[q] += az * cc[q];
          }
        }
      }
    }
    // feat row s: cols 0..7 = emb, 8..10 = pos, 11..31 = 0 (K padded to 32)
    #pragma unroll
    for (int q = 0; q < 4; ++q){
      featB[offF(s, 4*h + q)] = (_Float16)e[q];
      dembB[s*24 + 0*8 + 4*h + q] = dX[q] * scale;   // du/dp = scale
      dembB[s*24 + 1*8 + 4*h + q] = dY[q] * scale;
      dembB[s*24 + 2*8 + 4*h + q] = dZ[q] * scale;
    }
    if (h == 0){
      featB[offF(s, 8)]  = (_Float16)px;
      featB[offF(s, 9)]  = (_Float16)py;
      featB[offF(s, 10)] = (_Float16)pz;
      #pragma unroll
      for (int k = 11; k < 16; ++k) featB[offF(s, k)] = (_Float16)0.f;
    } else {
      const float4 z4 = make_float4(0.f, 0.f, 0.f, 0.f);
      *(float4*)(featB + offF(s, 16)) = z4;
      *(float4*)(featB + offF(s, 24)) = z4;
    }
  }

  // ---- cooperative weight staging: f32 global -> f16 padded LDS ----
  const int NT = 32 * WAVES_PER_BLOCK;
  for (int idx = tid; idx < 128*32; idx += NT){
    const int n = idx >> 5, k = idx & 31;
    const float v = (k < 11) ? W1g[k*128 + n] : 0.0f;
    sW1T[offF(n, k)] = (_Float16)v;
  }
  for (int idx = tid; idx < 16*128; idx += NT){
    const int k = idx >> 7, i = idx & 127;
    const float v = (k < 11) ? W1g[k*128 + i] : 0.0f;
    sW1[offW(k, i)] = (_Float16)v;
  }
  for (int idx = tid; idx < 128*128; idx += NT){
    const int i = idx >> 7, j = idx & 127;     // read W2 row-contiguous
    const float v = W2g[i*128 + j];
    sW2T[offW(j, i)] = (_Float16)v;
    sW2 [offW(i, j)] = (_Float16)v;
  }
  for (int t = tid; t < 128; t += NT){
    sW3[t] = W3g[t]; sB1[t] = b1g[t]; sB2[t] = b2g[t];
  }
  __syncthreads();

  const int mrow = lane & 15;
  const int hi   = lane >> 4;       // 0: rows 0..7 / k 0..7,16..23 of each K-32 step, 1: rows 8..15 / k 8..15,24..31
  const int rofs = hi * 8;

  // ================= L1 forward: z1 = feat @ W1 + b1 =================
  {
    FragU a;
    { const int k0 = hi << 3;
      a.q[0] = *(const float4*)(featB + offF(mrow, k0));
      a.q[1] = *(const float4*)(featB + offF(mrow, k0 + 16));
    }
    #pragma unroll
    for (int nt = 0; nt < 8; ++nt){
      const int col = nt*16 + mrow;
      FragU b;
      { const int k0 = hi << 3;
        b.q[0] = *(const float4*)(sW1T + offF(col, k0));
        b.q[1] = *(const float4*)(sW1T + offF(col, k0 + 16));
      }
      v8f acc; const float bias = sB1[col];
      #pragma unroll
      for (int r = 0; r < 8; ++r) acc[r] = bias;
      acc = wmma_f16(a, b, acc);
      #pragma unroll
      for (int r = 0; r < 8; ++r)
        z1B[offW(r + rofs, col)] = (_Float16)acc[r];
    }
  }

  // ================= L2 forward: z2 = sin(30 z1) @ W2 + b2 =================
  float sp[8];                       // per-lane partials of h2 . W3
  #pragma unroll
  for (int r = 0; r < 8; ++r) sp[r] = 0.f;
  {
    FragU A2[4];
    { const int kb = hi << 3;
      #pragma unroll
      for (int ks = 0; ks < 4; ++ks){
        FragU t;
        t.q[0] = *(const float4*)(z1B + offW(mrow, ks*32 + kb));
        t.q[1] = *(const float4*)(z1B + offW(mrow, ks*32 + kb + 16));
        #pragma unroll
        for (int e = 0; e < 16; ++e)
          t.h[e] = (_Float16)__sinf(30.f * (float)t.h[e]);   // h1
        A2[ks] = t;
      }
    }
    #pragma unroll
    for (int nt = 0; nt < 8; ++nt){
      const int col = nt*16 + mrow;
      v8f acc; const float bias = sB2[col];
      #pragma unroll
      for (int r = 0; r < 8; ++r) acc[r] = bias;
      #pragma unroll
      for (int ks = 0; ks < 4; ++ks){
        FragU b;
        const int k0 = ks*32 + (hi << 3);
        b.q[0] = *(const float4*)(sW2T + offW(col, k0));
        b.q[1] = *(const float4*)(sW2T + offW(col, k0 + 16));
        acc = wmma_f16(A2[ks], b, acc);
      }
      const float w3c = sW3[col];
      #pragma unroll
      for (int r = 0; r < 8; ++r){
        const float z2 = acc[r];
        sp[r] += __sinf(30.f * z2) * w3c;                       // h2 * W3
        gz2B[offW(r + rofs, col)] =
            (_Float16)(30.f * __cosf(30.f * z2) * w3c);         // gz2
      }
    }
  }

  // ---- sdf = h2.W3 + b3 : cross-lane reduce via LDS (feat region reused) ----
  #pragma unroll
  for (int r = 0; r < 8; ++r) redB[(r + rofs)*16 + mrow] = sp[r];
  if (lane < 16){
    float sdf = b3g[0];
    #pragma unroll
    for (int c = 0; c < 16; ++c) sdf += redB[lane*16 + c];
    outSdf[waveSample0 + lane] = sdf;
  }

  // ======== L2 backward: gh1 = gz2 @ W2^T ; gz1 = 30 cos(30 z1) * gh1 ========
  {
    FragU A3[4];
    { const int kb = hi << 3;
      #pragma unroll
      for (int ks = 0; ks < 4; ++ks){
        A3[ks].q[0] = *(const float4*)(gz2B + offW(mrow, ks*32 + kb));
        A3[ks].q[1] = *(const float4*)(gz2B + offW(mrow, ks*32 + kb + 16));
      }
    }
    #pragma unroll
    for (int nt = 0; nt < 8; ++nt){
      const int col = nt*16 + mrow;                 // h1-dim index i
      v8f acc;
      #pragma unroll
      for (int r = 0; r < 8; ++r) acc[r] = 0.f;
      #pragma unroll
      for (int ks = 0; ks < 4; ++ks){
        FragU b;                                    // B[j][i] = W2[i][j] = sW2 row i, contiguous in j
        const int k0 = ks*32 + (hi << 3);
        b.q[0] = *(const float4*)(sW2 + offW(col, k0));
        b.q[1] = *(const float4*)(sW2 + offW(col, k0 + 16));
        acc = wmma_f16(A3[ks], b, acc);
      }
      #pragma unroll
      for (int r = 0; r < 8; ++r){
        const int row = r + rofs;
        const float z1v = (float)z1B[offW(row, col)];
        z1B[offW(row, col)] =
            (_Float16)(30.f * __cosf(30.f * z1v) * acc[r]);     // gz1 in place
      }
    }
  }

  // ================= L1 backward: gfeat = gz1 @ W1^T =================
  {
    FragU A4[4];
    { const int kb = hi << 3;
      #pragma unroll
      for (int ks = 0; ks < 4; ++ks){
        A4[ks].q[0] = *(const float4*)(z1B + offW(mrow, ks*32 + kb));
        A4[ks].q[1] = *(const float4*)(z1B + offW(mrow, ks*32 + kb + 16));
      }
    }
    v8f acc;                                        // one N-tile: feat dims 0..15
    #pragma unroll
    for (int r = 0; r < 8; ++r) acc[r] = 0.f;
    #pragma unroll
    for (int ks = 0; ks < 4; ++ks){
      FragU b;                                      // B[i][k2] = W1[k2][i] = sW1 row k2, contiguous in i
      const int k0 = ks*32 + (hi << 3);
      b.q[0] = *(const float4*)(sW1 + offW(mrow, k0));
      b.q[1] = *(const float4*)(sW1 + offW(mrow, k0 + 16));
      acc = wmma_f16(A4[ks], b, acc);
    }
    // spill gfeat tile, then finish grad = gfeat[8:11] + J_emb^T gfeat[0:8]
    #pragma unroll
    for (int r = 0; r < 8; ++r) redB[(r + rofs)*16 + mrow] = acc[r];
    if (lane < 16){
      const int s = lane;
      const float* db = dembB + s*24;
      float g0 = redB[s*16 + 8], g1 = redB[s*16 + 9], g2 = redB[s*16 + 10];
      #pragma unroll
      for (int e = 0; e < 8; ++e){
        const float gf = redB[s*16 + e];
        g0 += gf * db[e]; g1 += gf * db[8 + e]; g2 += gf * db[16 + e];
      }
      const int gi = waveSample0 + s;
      outGrad[gi*3 + 0] = g0;
      outGrad[gi*3 + 1] = g1;
      outGrad[gi*3 + 2] = g2;
    }
  }
}

extern "C" void kernel_launch(void* const* d_in, const int* in_sizes, int n_in,
                              void* d_out, int out_size, void* d_ws, size_t ws_size,
                              hipStream_t stream) {
  const float* positions = (const float*)d_in[0];
  const float* encGrid   = (const float*)d_in[1];
  const unsigned char* mask = (const unsigned char*)d_in[2];
  const float* emptyGrid = (const float*)d_in[3];
  const float* W1 = (const float*)d_in[4];
  const float* b1 = (const float*)d_in[5];
  const float* W2 = (const float*)d_in[6];
  const float* b2 = (const float*)d_in[7];
  const float* W3 = (const float*)d_in[8];
  const float* b3 = (const float*)d_in[9];

  const int N = in_sizes[0] / 3;          // 262144
  float* outSdf  = (float*)d_out;
  float* outGrad = outSdf + N;

  const int blocks = N / (16 * WAVES_PER_BLOCK);   // 64 samples per 128-thread WG
  neural_sdf_fused<<<blocks, 32 * WAVES_PER_BLOCK, 0, stream>>>(
      positions, encGrid, mask, emptyGrid,
      W1, b1, W2, b2, W3, b3, outSdf, outGrad, N);
}
